// Generator_26396869001789
// MI455X (gfx1250) — matrix operand order repaired and checked by reading, not compiled, not run on hardware
//
#include <hip/hip_runtime.h>
#include <hip/hip_bf16.h>
#include <math.h>

typedef __bf16 bf16;
typedef bf16 v16bf __attribute__((ext_vector_type(16)));
typedef bf16 v8bf  __attribute__((ext_vector_type(8)));
typedef float v8f  __attribute__((ext_vector_type(8)));

#define NN 8192
#define DD 256
#define BN_EPS_F 1e-5f

union U32BF { unsigned u; bf16 b[2]; };
union Frag  { v16bf v; v8bf h[2]; };

// ---------------- edge scatter into bitmask adjacency ----------------
__global__ void k_scatter(const int* __restrict__ ei, unsigned* __restrict__ bits, int E) {
  int i = blockIdx.x * blockDim.x + threadIdx.x;
  if (i >= E) return;
  int s = ei[i];
  int d = ei[E + i];
  atomicOr(&bits[(size_t)s * (NN / 32) + (d >> 5)], 1u << (d & 31));
}

// ------- per-row sum of squared clipped adj_changes (struct_err^2), deterministic -------
__global__ void k_structsq(const float* __restrict__ adjch, float* __restrict__ structsq) {
  int tid = threadIdx.x;
  int row = blockIdx.x * 8 + (tid >> 5);
  int lane = tid & 31;
  const float* rp = adjch + (size_t)row * NN;
  float ss = 0.f;
  for (int j = lane; j < NN; j += 32) {
    float v = rp[j];
    v = (j == row) ? 0.f : fminf(1.f, fmaxf(-1.f, v));
    ss += v * v;
  }
#pragma unroll
  for (int off = 16; off > 0; off >>= 1) ss += __shfl_down(ss, off, 32);
  if (lane == 0) structsq[row] = ss;
}

// ------- build At = A_bin^T (bf16 0/1), tiled transpose through LDS -------
__global__ void k_buildAt(const float* __restrict__ adjch, const unsigned* __restrict__ bits,
                          bf16* __restrict__ At) {
  __shared__ bf16 T[64][65];
  int i0 = blockIdx.y * 64, j0 = blockIdx.x * 64;
  int tid = threadIdx.x;
  {
    int jl = tid & 63, il0 = tid >> 6;
#pragma unroll
    for (int p = 0; p < 16; ++p) {
      int il = il0 + p * 4;
      int i = i0 + il, j = j0 + jl;
      float v = adjch[(size_t)i * NN + j];
      v = (i == j) ? 0.f : fminf(1.f, fmaxf(-1.f, v));
      float a = v + (float)((bits[(size_t)i * (NN / 32) + (j >> 5)] >> (j & 31)) & 1u);
      T[il][jl] = (a != 0.f) ? (bf16)1.0f : (bf16)0.0f;
    }
  }
  __syncthreads();
  {
    int il = tid & 63, jl0 = tid >> 6;
#pragma unroll
    for (int p = 0; p < 16; ++p) {
      int jl = jl0 + p * 4;
      At[(size_t)(j0 + jl) * NN + i0 + il] = T[il][jl];
    }
  }
}

// ---------------- f32 -> bf16 converts ----------------
__global__ void k_cvt(const float* __restrict__ in, bf16* __restrict__ out, int n) {
  int i = blockIdx.x * 256 + threadIdx.x;
  if (i < n) out[i] = (bf16)in[i];
}
__global__ void k_cvt_t(const float* __restrict__ in, bf16* __restrict__ out, int R, int C) {
  // out[c][r] = in[r][c]  (R x C -> C x R)
  int i = blockIdx.x * 256 + threadIdx.x;
  if (i >= R * C) return;
  int r = i / C, c = i % C;
  out[(size_t)c * R + r] = (bf16)in[(size_t)r * C + c];
}

// ---- async direct-layout tile copy: 128 rows x 64B, global row stride ld (elems) ----
// Issues exactly 2 global_load_async_to_lds_b128 per thread (2 per wave counter-wise).
__device__ __forceinline__ void async_copy_tile(const bf16* __restrict__ g, int ld,
                                                unsigned ldsBase, int tid) {
#pragma unroll
  for (int q = 0; q < 2; ++q) {
    int qq = tid * 2 + q;     // 0..511 chunks of 16B
    int r = qq >> 2;          // row 0..127
    int c = qq & 3;           // 16B chunk within 64B row
    unsigned lds = ldsBase + (unsigned)(r * 80 + c * 16);
    unsigned long long src = (unsigned long long)(size_t)(g + (size_t)r * ld + c * 8);
    asm volatile("global_load_async_to_lds_b128 %0, %1, off"
                 :: "v"(lds), "v"(src) : "memory");
  }
}

// ---------------- generic bf16 WMMA GEMM, double-buffered async pipeline ----------------
// C[M x Nc] = A @ B' (+add)(+bias)
// A stored [M][K] (lda), direct async copy to LDS (double buffered).
// BT=true : B stored [Nc][K] (ldb = K-stride), direct async copy (double buffered).
// BT=false: B stored [K][Nc] (ldb), software-pipelined VGPR-staged transpose into LDS.
// Block tile 128x128, K-step 32, 8 waves (wave32), wave tile 32x64, 8 WMMAs/step.
template <bool BT, bool HAS_ADD, bool HAS_BIAS, bool WF, bool WB>
__global__ void k_gemm(const bf16* __restrict__ A, int lda,
                       const bf16* __restrict__ B, int ldb, int K,
                       const float* __restrict__ add, const float* __restrict__ bias,
                       float* __restrict__ outF, bf16* __restrict__ outB, int Nc) {
  __shared__ __align__(16) bf16 As[2][128][40]; // [m][k], 80B row stride keeps 16B align
  __shared__ __align__(16) bf16 Bs[2][128][40]; // [n][k]
  const int tid  = threadIdx.x;
  const int lane = tid & 31;
  const int wave = tid >> 5;
  const int wm = wave >> 1; // 0..3 -> M
  const int wn = wave & 1;  // 0..1 -> N
  const int bm = blockIdx.x * 128;
  const int bn = blockIdx.y * 128;
  const unsigned asB[2] = { (unsigned)(size_t)&As[0][0][0], (unsigned)(size_t)&As[1][0][0] };
  const unsigned bsB[2] = { (unsigned)(size_t)&Bs[0][0][0], (unsigned)(size_t)&Bs[1][0][0] };
  const int bc = tid & 63, bk0 = tid >> 6; // staged-B thread mapping

  v8f acc[2][4];
#pragma unroll
  for (int a = 0; a < 2; a++)
#pragma unroll
    for (int b = 0; b < 4; b++)
#pragma unroll
      for (int e = 0; e < 8; e++) acc[a][b][e] = 0.f;

  const int nt = K / 32;

  // ---- prologue: tile 0 into buffer 0
  async_copy_tile(A + (size_t)bm * lda, lda, asB[0], tid);
  if constexpr (BT) {
    async_copy_tile(B + (size_t)bn * ldb, ldb, bsB[0], tid);
  } else {
#pragma unroll
    for (int p = 0; p < 8; ++p) {
      int kl = bk0 + 4 * p;
      U32BF d; d.u = *(const unsigned*)(B + (size_t)kl * ldb + bn + 2 * bc);
      Bs[0][2 * bc][kl]     = d.b[0];
      Bs[0][2 * bc + 1][kl] = d.b[1];
    }
  }

  unsigned breg[8];
  for (int t = 0; t < nt; ++t) {
    const int cur = t & 1, nxt = cur ^ 1;
    const bool more = (t + 1) < nt;

    // ---- stage t+1 into nxt (WAR-safe: after barrier B of step t-1)
    if (more) {
      int kt1 = (t + 1) * 32;
      async_copy_tile(A + (size_t)bm * lda + kt1, lda, asB[nxt], tid);
      if constexpr (BT) {
        async_copy_tile(B + (size_t)bn * ldb + kt1, ldb, bsB[nxt], tid);
      } else {
#pragma unroll
        for (int p = 0; p < 8; ++p) {
          int kl = bk0 + 4 * p;
          breg[p] = *(const unsigned*)(B + (size_t)(kt1 + kl) * ldb + bn + 2 * bc);
        }
      }
      // block only until tile t's async copies have landed (t+1 stays in flight)
      if constexpr (BT) asm volatile("s_wait_asynccnt 0x4" ::: "memory");
      else              asm volatile("s_wait_asynccnt 0x2" ::: "memory");
    } else {
      asm volatile("s_wait_asynccnt 0x0" ::: "memory");
    }
    __syncthreads(); // sync A: tile t visible to all waves

    // ---- fragments (16-bit A 16x32 / B 32x16 lane layouts)
    Frag af[2], bfr[4];
    int khalf = (lane >> 4) * 8;
#pragma unroll
    for (int a = 0; a < 2; a++) {
      int m = wm * 32 + a * 16 + (lane & 15);
      af[a].h[0] = *(const v8bf*)&As[cur][m][khalf];      // K = half*8 .. +7
      af[a].h[1] = *(const v8bf*)&As[cur][m][16 + khalf]; // K = 16 + half*8 .. +7
    }
#pragma unroll
    for (int b = 0; b < 4; b++) {
      int n = wn * 64 + b * 16 + (lane & 15);
      bfr[b].h[0] = *(const v8bf*)&Bs[cur][n][khalf];
      bfr[b].h[1] = *(const v8bf*)&Bs[cur][n][16 + khalf];
    }
#pragma unroll
    for (int a = 0; a < 2; a++)
#pragma unroll
      for (int b = 0; b < 4; b++)
        acc[a][b] = __builtin_amdgcn_wmma_f32_16x16x32_bf16(
            false, af[a].v, false, bfr[b].v, (short)0, acc[a][b], false, false);

    // ---- drain staged B for t+1 after compute (global latency hidden under WMMAs)
    if (!BT && more) {
#pragma unroll
      for (int p = 0; p < 8; ++p) {
        int kl = bk0 + 4 * p;
        U32BF d; d.u = breg[p];
        Bs[nxt][2 * bc][kl]     = d.b[0];
        Bs[nxt][2 * bc + 1][kl] = d.b[1];
      }
    }
    __syncthreads(); // sync B: all waves done reading tile t; cur reusable at t+1
  }

  // ---- branch-free epilogue: C layout m = e + (lane>=16)*8, n = lane&15
#pragma unroll
  for (int a = 0; a < 2; a++) {
#pragma unroll
    for (int b = 0; b < 4; b++) {
      int n = bn + wn * 64 + b * 16 + (lane & 15);
#pragma unroll
      for (int e = 0; e < 8; e++) {
        int m = bm + wm * 32 + a * 16 + (lane >> 4) * 8 + e;
        float v = acc[a][b][e];
        if constexpr (HAS_ADD)  v += add[(size_t)m * Nc + n];
        if constexpr (HAS_BIAS) v += bias[n];
        if constexpr (WF) outF[(size_t)m * Nc + n] = v;
        if constexpr (WB) outB[(size_t)m * Nc + n] = (bf16)v;
      }
    }
  }
}

// ---------------- BatchNorm stats (deterministic two-stage) ----------------
__global__ void k_bnstat1(const float* __restrict__ h2, float* __restrict__ pS, float* __restrict__ pQ) {
  int colg = blockIdx.x, chunk = blockIdx.y;
  int lane = threadIdx.x & 31, rg = threadIdx.x >> 5;
  int col = colg * 32 + lane;
  float s = 0.f, q = 0.f;
  int r0 = chunk * 256;
  for (int r = r0 + rg; r < r0 + 256; r += 8) {
    float v = h2[(size_t)r * DD + col];
    s += v; q += v * v;
  }
  __shared__ float S[8][32], Q[8][32];
  S[rg][lane] = s; Q[rg][lane] = q;
  __syncthreads();
  if (rg == 0) {
#pragma unroll
    for (int i = 1; i < 8; i++) { s += S[i][lane]; q += Q[i][lane]; }
    pS[chunk * DD + col] = s; pQ[chunk * DD + col] = q;
  }
}
__global__ void k_bnstat2(const float* __restrict__ pS, const float* __restrict__ pQ,
                          const float* __restrict__ gamma, const float* __restrict__ beta,
                          float* __restrict__ scale, float* __restrict__ shift) {
  int col = threadIdx.x;
  float s = 0.f, q = 0.f;
  for (int c = 0; c < 32; c++) { s += pS[c * DD + col]; q += pQ[c * DD + col]; }
  float mean = s / (float)NN;
  float var  = q / (float)NN - mean * mean;
  float inv  = rsqrtf(var + BN_EPS_F);
  float sc   = gamma[col] * inv;
  scale[col] = sc;
  shift[col] = beta[col] - mean * sc;
}
__global__ void k_bnrelu(const float* __restrict__ h2, const float* __restrict__ scale,
                         const float* __restrict__ shift, float* __restrict__ hF,
                         bf16* __restrict__ hB) {
  int i = blockIdx.x * 256 + threadIdx.x;
  int c = i & (DD - 1);
  float v = fmaxf(0.f, h2[i] * scale[c] + shift[c]);
  hF[i] = v;
  hB[i] = (bf16)v;
}

// ---------------- head: sigmoid + per-row score ----------------
__global__ void k_score(const float* __restrict__ hF, const float* __restrict__ x,
                        const float* __restrict__ mlpW, const float* __restrict__ mlpb,
                        const float* __restrict__ structsq,
                        float* __restrict__ out, float* __restrict__ score) {
  int tid = threadIdx.x;
  int row = blockIdx.x * 8 + (tid >> 5);
  int lane = tid & 31;
  const float* hr = hF + (size_t)row * DD;
  const float* xr = x + (size_t)row * DD;
  float dot = 0.f, sx = 0.f, sxx = 0.f;
  for (int f = lane; f < DD; f += 32) {
    dot += hr[f] * mlpW[f];
    float xv = xr[f];
    sx += xv; sxx += xv * xv;
  }
#pragma unroll
  for (int off = 16; off > 0; off >>= 1) {
    dot += __shfl_down(dot, off, 32);
    sx  += __shfl_down(sx,  off, 32);
    sxx += __shfl_down(sxx, off, 32);
  }
  if (lane == 0) {
    float z = dot + mlpb[0];
    float o = 1.f / (1.f + expf(-z));
    out[row] = o;
    float attr = sqrtf(fmaxf(0.f, (float)DD * o * o - 2.f * o * sx + sxx));
    score[row] = 0.5f * attr + 0.5f * sqrtf(structsq[row]);
  }
}
__global__ void k_mean(const float* __restrict__ score, float* __restrict__ outScalar) {
  int t = threadIdx.x;
  float s = 0.f;
  for (int i = t; i < NN; i += 256) s += score[i];
  __shared__ float S[256];
  S[t] = s;
  __syncthreads();
  for (int off = 128; off > 0; off >>= 1) {
    if (t < off) S[t] += S[t + off];
    __syncthreads();
  }
  if (t == 0) outScalar[0] = S[0] / (float)NN;
}

extern "C" void kernel_launch(void* const* d_in, const int* in_sizes, int n_in,
                              void* d_out, int out_size, void* d_ws, size_t ws_size,
                              hipStream_t stream) {
  const float* x     = (const float*)d_in[0];
  const int*   ei    = (const int*)d_in[1];
  /* d_in[2] latent: unused by reference */
  const float* adjch = (const float*)d_in[3];
  const float* fc    = (const float*)d_in[4];
  const float* ftW   = (const float*)d_in[5];
  const float* ftb   = (const float*)d_in[6];
  const float* gnnW  = (const float*)d_in[7];
  const float* gnnb  = (const float*)d_in[8];
  const float* gamma = (const float*)d_in[9];
  const float* beta  = (const float*)d_in[10];
  const float* mlpW  = (const float*)d_in[11];
  const float* mlpb  = (const float*)d_in[12];
  float* out = (float*)d_out;
  const int E = in_sizes[1] / 2;

  char* w = (char*)d_ws;
  auto alloc = [&](size_t bytes) -> char* {
    char* p = w;
    w += (bytes + 255) & ~(size_t)255;
    return p;
  };
  bf16*     At    = (bf16*)alloc((size_t)NN * NN * 2);      // A_bin^T, 128 MB (L2-resident)
  unsigned* bits  = (unsigned*)alloc((size_t)NN * NN / 8);  // 8 MB
  bf16*     xb    = (bf16*)alloc((size_t)NN * DD * 2);
  bf16*     fcT   = (bf16*)alloc((size_t)DD * DD * 2);      // feature_change^T  [n][k]
  bf16*     ftWb  = (bf16*)alloc((size_t)DD * DD * 2);      // ft_W natural = [n][k]
  bf16*     gWb   = (bf16*)alloc((size_t)3 * DD * DD * 2);  // gnn_W natural = [n][k]
  bf16*     g0b   = (bf16*)alloc((size_t)NN * DD * 2);
  float*    hF    = (float*)alloc((size_t)NN * DD * 4);
  bf16*     hB    = (bf16*)alloc((size_t)NN * DD * 2);
  bf16*     tmpB  = (bf16*)alloc((size_t)NN * DD * 2);
  float*    h2    = (float*)alloc((size_t)NN * DD * 4);
  float*    pS    = (float*)alloc((size_t)32 * DD * 4);
  float*    pQ    = (float*)alloc((size_t)32 * DD * 4);
  float*    scl   = (float*)alloc((size_t)DD * 4);
  float*    shf   = (float*)alloc((size_t)DD * 4);
  float*    ssq   = (float*)alloc((size_t)NN * 4);
  float*    score = (float*)alloc((size_t)NN * 4);
  (void)ws_size; (void)n_in; (void)out_size;

  hipMemsetAsync(bits, 0, (size_t)NN * NN / 8, stream);
  k_scatter<<<(E + 255) / 256, 256, 0, stream>>>(ei, bits, E);
  k_structsq<<<NN / 8, 256, 0, stream>>>(adjch, ssq);
  k_buildAt<<<dim3(NN / 64, NN / 64), 256, 0, stream>>>(adjch, bits, At);
  k_cvt<<<(NN * DD) / 256, 256, 0, stream>>>(x, xb, NN * DD);
  k_cvt<<<(DD * DD) / 256, 256, 0, stream>>>(ftW, ftWb, DD * DD);
  k_cvt<<<(3 * DD * DD) / 256, 256, 0, stream>>>(gnnW, gWb, 3 * DD * DD);
  k_cvt_t<<<(DD * DD) / 256, 256, 0, stream>>>(fc, fcT, DD, DD);

  dim3 gg(NN / 128, DD / 128);
  // g0 = x @ feature_change            (B' = FC^T, direct)
  k_gemm<true, false, false, false, true><<<gg, 256, 0, stream>>>(
      xb, DD, fcT, DD, DD, nullptr, nullptr, nullptr, g0b, DD);
  // h = g0 @ ft_W^T + ft_b             (B' = ft_W natural)
  k_gemm<true, false, true, true, true><<<gg, 256, 0, stream>>>(
      g0b, DD, ftWb, DD, DD, nullptr, ftb, hF, hB, DD);

  for (int l = 0; l < 3; l++) {
    // tmp = A_bin^T @ h + h   (dominant: A = At async double-buffered, B = h staged)
    k_gemm<false, true, false, false, true><<<gg, 256, 0, stream>>>(
        At, NN, hB, DD, NN, hF, nullptr, nullptr, tmpB, DD);
    // h2 = tmp @ gnn_W[l]^T + gnn_b[l]  (B' = gnn_W natural)
    k_gemm<true, false, true, true, false><<<gg, 256, 0, stream>>>(
        tmpB, DD, gWb + l * DD * DD, DD, DD, nullptr, gnnb + l * DD, h2, nullptr, DD);
    k_bnstat1<<<dim3(8, 32), 256, 0, stream>>>(h2, pS, pQ);
    k_bnstat2<<<1, 256, 0, stream>>>(pS, pQ, gamma + l * DD, beta + l * DD, scl, shf);
    k_bnrelu<<<(NN * DD) / 256, 256, 0, stream>>>(h2, scl, shf, hF, hB);
  }

  k_score<<<NN / 8, 256, 0, stream>>>(hF, x, mlpW, mlpb, ssq, out, score);
  k_mean<<<1, 256, 0, stream>>>(score, out + NN);
}
